// VQVAE_41162966565649
// MI455X (gfx1250) — matrix operand order repaired and checked by reading, not compiled, not run on hardware
//
#include <hip/hip_runtime.h>
#include <hip/hip_bf16.h>

typedef __attribute__((ext_vector_type(16))) _Float16 v16h;
typedef __attribute__((ext_vector_type(8)))  float    v8f;

// ---------------------------------------------------------------------------
// Block-cooperative implicit-GEMM convolution (NCHW / OIHW) using WMMA.
//   A = W [Cout x K] (K = Cin*Kh*Kw contiguous),  B = im2col(X) [K x N].
// Block (256 thr = 8 waves) owns a 128(M) x 128(N) tile; wave (wm,wn) owns
// 32x64.  Per 32-wide K-step the block stages B (32x128, f16) into LDS once
// (double-buffered, one barrier per step); each wave then builds 4 B-fragments
// with aligned ds reads and issues 8 v_wmma_f32_16x16x32_f16.
// Fused bias / relu_in / relu_out / residual-add.
// Requirements (all satisfied by the launches below):
//   K % 16 == 0, H/W dims are powers of two, Ntot % 128 == 0.
// No early returns: barriers are safe, EXEC all-ones at every WMMA.
// ---------------------------------------------------------------------------
__global__ __launch_bounds__(256) void conv_wmma_kernel(
    const float* __restrict__ X, const float* __restrict__ W,
    const float* __restrict__ Bias, const float* __restrict__ Res,
    float* __restrict__ Y,
    int Bn, int Cin, int lHin, int lWin,
    int Cout, int lHout, int lWout,
    int Kh, int Kw, int stride, int pad,
    int relu_in, int relu_out)
{
  __shared__ _Float16 sB[2][128][40];   // [stage][n_local][k], padded stride

  const int Hin  = 1 << lHin,  Win  = 1 << lWin;
  const int Hout = 1 << lHout, Wout = 1 << lWout;
  const int Ntot = Bn << (lHout + lWout);
  const int K    = Cin * Kh * Kw;
  const int KhKw = Kh * Kw;
  const int HW   = Hin * Win;

  const int tid  = threadIdx.x;
  const int lane = tid & 31;
  const int wid  = tid >> 5;
  const int wm   = wid & 3;        // 4 M-waves share the staged B tile
  const int wn   = wid >> 2;       // 2 N-waves
  const int m0    = blockIdx.y * 128 + wm * 32;
  const int n0blk = blockIdx.x * 128;
  const int n0    = n0blk + wn * 64;

  const int hi = lane >> 4;        // half-wave selector per ISA fragment tables
  const int la = lane & 15;

  // ---- staging role: thread stages column nl = tid&127, k-span (tid>>7)*16.
  const int snl = tid & 127;
  const int sk0 = (tid >> 7) * 16;
  int s_ihb, s_iwb; long long s_xbb;
  {
    const int n  = n0blk + snl;
    const bool ok = n < Ntot;
    const int ow = n & (Wout - 1);
    const int oh = (n >> lWout) & (Hout - 1);
    const int bb = n >> (lWout + lHout);
    s_ihb = ok ? (oh * stride - pad) : -(1 << 20);   // mask via bounds check
    s_iwb = ow * stride - pad;
    s_xbb = (long long)bb * Cin * HW;
  }

  // ---- compute role: A rows (two 16-row subtiles) ----
  int  am[2]; bool am_ok[2]; long long arow[2];
#pragma unroll
  for (int mi = 0; mi < 2; ++mi) {
    am[mi]    = m0 + mi * 16 + la;
    am_ok[mi] = am[mi] < Cout;
    arow[mi]  = (long long)am[mi] * K;
  }

  // ---- compute role: output pixel addressing for stores ----
  bool n_ok[4]; long long pixbase[4];
  const long long HWo = (long long)Hout * Wout;
#pragma unroll
  for (int sub = 0; sub < 4; ++sub) {
    const int n = n0 + sub * 16 + la;
    n_ok[sub] = n < Ntot;
    const int ow = n & (Wout - 1);
    const int oh = (n >> lWout) & (Hout - 1);
    const int bb = n >> (lWout + lHout);
    pixbase[sub] = (long long)bb * Cout * HWo + (long long)oh * Wout + ow;
  }

  v8f acc[2][4] = {};
  const int nsteps = (K + 31) >> 5;

  auto stage = [&](int k0, int buf) {
    int k = k0 + sk0;
    unsigned ci = (unsigned)k / (unsigned)KhKw;
    unsigned rr = (unsigned)k % (unsigned)KhKw;
    int kh = (int)(rr / (unsigned)Kw);
    int kw = (int)(rr % (unsigned)Kw);
    _Float16* dst = &sB[buf][snl][sk0];
#pragma unroll
    for (int j = 0; j < 16; ++j) {       // k >= K  <=>  ci >= Cin
      float val = 0.0f;
      if ((int)ci < Cin) {
        const int ih = s_ihb + kh, iw = s_iwb + kw;
        if ((unsigned)ih < (unsigned)Hin && (unsigned)iw < (unsigned)Win) {
          val = X[s_xbb + (long long)ci * HW + (ih << lWin) + iw];
          if (relu_in) val = fmaxf(val, 0.0f);
        }
      }
      dst[j] = (_Float16)val;
      if (++kw == Kw) { kw = 0; if (++kh == Kh) { kh = 0; ++ci; } }
    }
  };

  stage(0, 0);
  for (int s = 0; s < nsteps; ++s) {
    const int k0 = s << 5;
    __syncthreads();                      // stage s visible; stage s-1 reads done
    if (s + 1 < nsteps) stage(k0 + 32, (s + 1) & 1);

    // A fragments: two contiguous 8-float runs per fragment, 32B-aligned.
    v16h a[2];
#pragma unroll
    for (int mi = 0; mi < 2; ++mi) {
      float av[16];
#pragma unroll
      for (int r = 0; r < 2; ++r) {
        const int kb = k0 + 16 * hi + 8 * r;
        if (am_ok[mi] && kb < K) {
          const float4 p0 = *(const float4*)(W + arow[mi] + kb);
          const float4 p1 = *(const float4*)(W + arow[mi] + kb + 4);
          av[8*r+0] = p0.x; av[8*r+1] = p0.y; av[8*r+2] = p0.z; av[8*r+3] = p0.w;
          av[8*r+4] = p1.x; av[8*r+5] = p1.y; av[8*r+6] = p1.z; av[8*r+7] = p1.w;
        } else {
#pragma unroll
          for (int j = 0; j < 8; ++j) av[8*r+j] = 0.0f;
        }
      }
#pragma unroll
      for (int e = 0; e < 16; ++e) a[mi][e] = (_Float16)av[e];
    }

    // B fragments from LDS: element e needs k = k0 + 16*hi + e -> contiguous.
    const int buf = s & 1;
#pragma unroll
    for (int sub = 0; sub < 4; ++sub) {
      const _Float16* p = &sB[buf][wn * 64 + sub * 16 + la][hi * 16];
      v16h bf;
#pragma unroll
      for (int e = 0; e < 16; ++e) bf[e] = p[e];
#pragma unroll
      for (int mi = 0; mi < 2; ++mi)
        acc[mi][sub] = __builtin_amdgcn_wmma_f32_16x16x32_f16(
            false, a[mi], false, bf, (short)0, acc[mi][sub], false, false);
    }
  }

  // ---- store D: lane holds rows m = mi*16 + 8*hi + r, col = la of subtile.
  const bool hasRes = (Res != nullptr);
  float bias_v[2][8];
#pragma unroll
  for (int mi = 0; mi < 2; ++mi)
#pragma unroll
    for (int r = 0; r < 8; ++r) {
      const int cout = m0 + mi * 16 + 8 * hi + r;
      bias_v[mi][r] = (cout < Cout) ? Bias[cout] : 0.0f;
    }
#pragma unroll
  for (int sub = 0; sub < 4; ++sub) {
    if (!n_ok[sub]) continue;
#pragma unroll
    for (int mi = 0; mi < 2; ++mi)
#pragma unroll
      for (int r = 0; r < 8; ++r) {
        const int cout = m0 + mi * 16 + 8 * hi + r;
        if (cout < Cout) {
          const long long idx = pixbase[sub] + (long long)cout * HWo;
          float v = acc[mi][sub][r] + bias_v[mi][r];
          if (hasRes) v += Res[idx];
          if (relu_out) v = fmaxf(v, 0.0f);
          Y[idx] = v;
        }
      }
  }
}

// ---------------------------------------------------------------------------
// ConvTranspose2d(k=4, s=2, p=1) as implicit GEMM (torch semantics):
//   y[b,co,oh,ow] = sum_{ci,kh,kw} x[b,ci,(oh+1-kh)/2,(ow+1-kw)/2] * w[ci,co,kh,kw]
// Weights (Cin, Cout, 4, 4), K = Cin*16.  32x64 wave tile, 8 WMMAs per step.
// (Private gathers; Cout here is 64 / 3, so LDS-cooperative staging would
// mostly idle waves.)
// ---------------------------------------------------------------------------
__global__ __launch_bounds__(128) void convT_wmma_kernel(
    const float* __restrict__ X, const float* __restrict__ Wt,
    const float* __restrict__ Bias, float* __restrict__ Y,
    int Bn, int Cin, int Hin, int Win, int Cout, int relu_out)
{
  const int Hout = Hin * 2, Wout = Win * 2;
  const int lane = threadIdx.x & 31;
  const int wid  = threadIdx.x >> 5;
  const int Ntot   = Bn * Hout * Wout;
  const int tilesM = (Cout + 31) >> 5;
  const int tilesN = (Ntot + 63) >> 6;
  const int tile   = blockIdx.x * 4 + wid;
  if (tile >= tilesM * tilesN) return;            // no barriers in this kernel
  const int tm = tile % tilesM, tn = tile / tilesM;
  const int m0 = tm * 32,       n0 = tn * 64;
  const int K  = Cin * 16;
  const int HW = Hin * Win;

  const int hi = lane >> 4, la = lane & 15;

  int  am[2]; bool am_ok[2];
#pragma unroll
  for (int mi = 0; mi < 2; ++mi) {
    am[mi]    = m0 + mi * 16 + la;
    am_ok[mi] = am[mi] < Cout;
  }

  int ohv[4], owv[4]; bool n_ok[4];
  long long xbb[4], pixbase[4];
  const long long HWo = (long long)Hout * Wout;
#pragma unroll
  for (int sub = 0; sub < 4; ++sub) {
    const int n = n0 + sub * 16 + la;
    n_ok[sub] = n < Ntot;
    int ow = 0, oh = 0, bb = 0;
    if (n_ok[sub]) { ow = n % Wout; int t = n / Wout; oh = t % Hout; bb = t / Hout; }
    ohv[sub] = n_ok[sub] ? oh : -(1 << 20);
    owv[sub] = ow;
    xbb[sub] = (long long)bb * Cin * HW;
    pixbase[sub] = (long long)bb * Cout * HWo + (long long)oh * Wout + ow;
  }

  v8f acc[2][4] = {};

  for (int k0 = 0; k0 < K; k0 += 32) {
    v16h a[2];
#pragma unroll
    for (int mi = 0; mi < 2; ++mi) {
      float av[16];
#pragma unroll
      for (int r = 0; r < 2; ++r) {
        const int kb = k0 + 16 * hi + 8 * r;
        if (am_ok[mi] && kb < K) {
          const int ci = kb >> 4, kk = kb & 15;
          const float* wp = Wt + ((long long)ci * Cout + am[mi]) * 16 + kk;
          const float4 p0 = *(const float4*)(wp);
          const float4 p1 = *(const float4*)(wp + 4);
          av[8*r+0] = p0.x; av[8*r+1] = p0.y; av[8*r+2] = p0.z; av[8*r+3] = p0.w;
          av[8*r+4] = p1.x; av[8*r+5] = p1.y; av[8*r+6] = p1.z; av[8*r+7] = p1.w;
        } else {
#pragma unroll
          for (int j = 0; j < 8; ++j) av[8*r+j] = 0.0f;
        }
      }
#pragma unroll
      for (int e = 0; e < 16; ++e) a[mi][e] = (_Float16)av[e];
    }

    const int kb = k0 + 16 * hi;   // 16 contiguous k == exactly one ci
    const int ci = kb >> 4;
    const bool ci_ok = kb < K;

#pragma unroll
    for (int sub = 0; sub < 4; ++sub) {
      v16h bf;
#pragma unroll
      for (int j = 0; j < 16; ++j) {
        const int kh = j >> 2, kw = j & 3;
        float val = 0.0f;
        if (ci_ok) {
          const int ih2 = ohv[sub] + 1 - kh, iw2 = owv[sub] + 1 - kw;
          if (ih2 >= 0 && iw2 >= 0 && !(ih2 & 1) && !(iw2 & 1)) {
            const int ih = ih2 >> 1, iw = iw2 >> 1;
            if (ih < Hin && iw < Win)
              val = X[xbb[sub] + (long long)ci * HW + ih * Win + iw];
          }
        }
        bf[j] = (_Float16)val;
      }
#pragma unroll
      for (int mi = 0; mi < 2; ++mi)
        acc[mi][sub] = __builtin_amdgcn_wmma_f32_16x16x32_f16(
            false, a[mi], false, bf, (short)0, acc[mi][sub], false, false);
    }
  }

  float bias_v[2][8];
#pragma unroll
  for (int mi = 0; mi < 2; ++mi)
#pragma unroll
    for (int r = 0; r < 8; ++r) {
      const int cout = m0 + mi * 16 + 8 * hi + r;
      bias_v[mi][r] = (cout < Cout) ? Bias[cout] : 0.0f;
    }
#pragma unroll
  for (int sub = 0; sub < 4; ++sub) {
    if (!n_ok[sub]) continue;
#pragma unroll
    for (int mi = 0; mi < 2; ++mi)
#pragma unroll
      for (int r = 0; r < 8; ++r) {
        const int cout = m0 + mi * 16 + 8 * hi + r;
        if (cout < Cout) {
          float v = acc[mi][sub][r] + bias_v[mi][r];
          if (relu_out) v = fmaxf(v, 0.0f);
          Y[pixbase[sub] + (long long)cout * HWo] = v;
        }
      }
  }
}

// ---------------------------------------------------------------------------
// Vector quantizer: per pixel n (65536 total), nearest of 512 codes (D=32).
// Codebook staged in dynamic LDS (64 KB + 1 KB reduce; WGP has 320 KB).
// best distance == ||q - z||^2, so vq_loss = 1.25 * sum(best)/(65536*32).
// Z/Q are NCHW [64,32,32,32]: element (b,d,p) at b*32768 + d*1024 + p.
// ---------------------------------------------------------------------------
__global__ __launch_bounds__(256) void vq_kernel(
    const float* __restrict__ Z, const float* __restrict__ Emb,
    float* __restrict__ Q, float* __restrict__ lossAccum)
{
  extern __shared__ float smem[];          // [512*32] codebook + [256] reduce
  float* se  = smem;
  float* red = smem + 512 * 32;
  for (int i = threadIdx.x; i < 512 * 32; i += 256) se[i] = Emb[i];
  __syncthreads();

  const int n   = blockIdx.x * 256 + threadIdx.x;   // grid covers 65536 exactly
  const int b   = n >> 10, rem = n & 1023;
  const long long base = (long long)b * 32768 + rem;

  float z[32];
#pragma unroll
  for (int d = 0; d < 32; ++d) z[d] = Z[base + d * 1024];

  float best = 3.4e38f; int bi = 0;
  for (int k = 0; k < 512; ++k) {
    const float* e = se + k * 32;
    float s = 0.0f;
#pragma unroll
    for (int d = 0; d < 32; ++d) { const float df = z[d] - e[d]; s = fmaf(df, df, s); }
    if (s < best) { best = s; bi = k; }   // strict < => first argmin, like jnp
  }
#pragma unroll
  for (int d = 0; d < 32; ++d) Q[base + d * 1024] = se[bi * 32 + d];

  red[threadIdx.x] = best;
  __syncthreads();
  for (int s = 128; s > 0; s >>= 1) {
    if ((int)threadIdx.x < s) red[threadIdx.x] += red[threadIdx.x + s];
    __syncthreads();
  }
  if (threadIdx.x == 0) atomicAdd(lossAccum, red[0]);
}

__global__ void zero_kernel(float* p) { p[0] = 0.0f; }

__global__ void finalize_loss_kernel(const float* __restrict__ accum,
                                     float* __restrict__ out) {
  out[0] = 1.25f * accum[0] / (65536.0f * 32.0f);
}

// ---------------------------------------------------------------------------
extern "C" void kernel_launch(void* const* d_in, const int* in_sizes, int n_in,
                              void* d_out, int out_size, void* d_ws, size_t ws_size,
                              hipStream_t stream) {
  (void)in_sizes; (void)n_in; (void)out_size; (void)ws_size;

  const float* x      = (const float*)d_in[0];
  const float* ec0_w  = (const float*)d_in[1];   const float* ec0_b  = (const float*)d_in[2];
  const float* ec1_w  = (const float*)d_in[3];   const float* ec1_b  = (const float*)d_in[4];
  const float* er1a_w = (const float*)d_in[5];   const float* er1a_b = (const float*)d_in[6];
  const float* er1b_w = (const float*)d_in[7];   const float* er1b_b = (const float*)d_in[8];
  const float* er2a_w = (const float*)d_in[9];   const float* er2a_b = (const float*)d_in[10];
  const float* er2b_w = (const float*)d_in[11];  const float* er2b_b = (const float*)d_in[12];
  const float* ec2_w  = (const float*)d_in[13];  const float* ec2_b  = (const float*)d_in[14];
  const float* emb    = (const float*)d_in[15];
  const float* dc0_w  = (const float*)d_in[16];  const float* dc0_b  = (const float*)d_in[17];
  const float* dr1a_w = (const float*)d_in[18];  const float* dr1a_b = (const float*)d_in[19];
  const float* dr1b_w = (const float*)d_in[20];  const float* dr1b_b = (const float*)d_in[21];
  const float* dr2a_w = (const float*)d_in[22];  const float* dr2a_b = (const float*)d_in[23];
  const float* dr2b_w = (const float*)d_in[24];  const float* dr2b_b = (const float*)d_in[25];
  const float* dt1_w  = (const float*)d_in[26];  const float* dt1_b  = (const float*)d_in[27];
  const float* dt2_w  = (const float*)d_in[28];  const float* dt2_b  = (const float*)d_in[29];

  float* out = (float*)d_out;                    // [64,3,128,128] + loss scalar
  char*  ws  = (char*)d_ws;
  // Workspace layout (~96 MB):
  //   bufA  @ 0         : 67108864 B  ([64,64,64,64]; later dt1 output)
  //   bufB  @ 67108864  : 33554432 B  ([64,128,32,32] main stream)
  //   accum @ 100663296 : 4 B
  // Aliases inside bufA (lifetimes disjoint with bufA's own uses):
  //   bufC = bufA[0 .. 33.5MB)      res-block temp ([64,128,32,32])
  //   bufZ = ws+33554432 (8.4 MB)   z  ([64,32,32,32])
  //   bufQ = ws+41943040 (8.4 MB)   q  ([64,32,32,32])
  float* bufA  = (float*)(ws);
  float* bufB  = (float*)(ws + 67108864);
  float* accum = (float*)(ws + 100663296);
  float* bufC  = bufA;
  float* bufZ  = (float*)(ws + 33554432);
  float* bufQ  = (float*)(ws + 41943040);

  auto lg = [](int v) { int l = 0; while ((1 << l) < v) ++l; return l; };

  auto conv = [&](const float* X, const float* Wp, const float* Bi, const float* Rs,
                  float* Yp, int Bn, int Cin, int Hin, int Win,
                  int Cout, int Hout, int Wout, int Kh, int Kw, int s, int p,
                  int ri, int ro) {
    const int Ntot   = Bn * Hout * Wout;
    const int tilesN = (Ntot + 127) / 128;
    const int tilesM = (Cout + 127) / 128;
    dim3 grid(tilesN, tilesM);
    conv_wmma_kernel<<<grid, 256, 0, stream>>>(
        X, Wp, Bi, Rs, Yp, Bn, Cin, lg(Hin), lg(Win),
        Cout, lg(Hout), lg(Wout), Kh, Kw, s, p, ri, ro);
  };
  auto convT = [&](const float* X, const float* Wp, const float* Bi, float* Yp,
                   int Bn, int Cin, int Hin, int Win, int Cout, int ro) {
    const int Ntot  = Bn * (Hin * 2) * (Win * 2);
    const int tiles = ((Cout + 31) / 32) * ((Ntot + 63) / 64);
    convT_wmma_kernel<<<(tiles + 3) / 4, 128, 0, stream>>>(
        X, Wp, Bi, Yp, Bn, Cin, Hin, Win, Cout, ro);
  };

  // ---------------- encoder ----------------
  conv(x,    ec0_w, ec0_b, nullptr, bufA, 64,   3, 128, 128,  64, 64, 64, 4, 4, 2, 1, 0, 1);
  conv(bufA, ec1_w, ec1_b, nullptr, bufB, 64,  64,  64,  64, 128, 32, 32, 4, 4, 2, 1, 0, 1);
  // res block 1: bufC = conv3x3(relu(bufB)); bufB += conv1x1(relu(bufC))
  conv(bufB, er1a_w, er1a_b, nullptr, bufC, 64, 128, 32, 32, 128, 32, 32, 3, 3, 1, 1, 1, 0);
  conv(bufC, er1b_w, er1b_b, bufB,    bufB, 64, 128, 32, 32, 128, 32, 32, 1, 1, 1, 0, 1, 0);
  // res block 2
  conv(bufB, er2a_w, er2a_b, nullptr, bufC, 64, 128, 32, 32, 128, 32, 32, 3, 3, 1, 1, 1, 0);
  conv(bufC, er2b_w, er2b_b, bufB,    bufB, 64, 128, 32, 32, 128, 32, 32, 1, 1, 1, 0, 1, 0);
  // z = conv3x3 -> [64,32,32,32]
  conv(bufB, ec2_w, ec2_b, nullptr, bufZ, 64, 128, 32, 32, 32, 32, 32, 3, 3, 1, 1, 0, 0);

  // ---------------- vector quantizer ----------------
  zero_kernel<<<1, 1, 0, stream>>>(accum);
  vq_kernel<<<256, 256, (512 * 32 + 256) * sizeof(float), stream>>>(bufZ, emb, bufQ, accum);
  finalize_loss_kernel<<<1, 1, 0, stream>>>(accum, out + 3145728);

  // ---------------- decoder ----------------
  conv(bufQ, dc0_w, dc0_b, nullptr, bufB, 64,  32, 32, 32, 128, 32, 32, 3, 3, 1, 1, 0, 0);
  conv(bufB, dr1a_w, dr1a_b, nullptr, bufC, 64, 128, 32, 32, 128, 32, 32, 3, 3, 1, 1, 1, 0);
  conv(bufC, dr1b_w, dr1b_b, bufB,    bufB, 64, 128, 32, 32, 128, 32, 32, 1, 1, 1, 0, 1, 0);
  conv(bufB, dr2a_w, dr2a_b, nullptr, bufC, 64, 128, 32, 32, 128, 32, 32, 3, 3, 1, 1, 1, 0);
  conv(bufC, dr2b_w, dr2b_b, bufB,    bufB, 64, 128, 32, 32, 128, 32, 32, 1, 1, 1, 0, 1, 0);
  convT(bufB, dt1_w, dt1_b, bufA, 64, 128, 32, 32, 64, 1);   // relu
  convT(bufA, dt2_w, dt2_b, out,  64,  64, 64, 64,  3, 0);   // -> d_out
}
